// EnhancedEncoderLayer_88983132438635
// MI455X (gfx1250) — matrix operand order repaired and checked by reading, not compiled, not run on hardware
//
#include <hip/hip_runtime.h>
#include <hip/hip_bf16.h>
#include <math.h>

// ---------------------------------------------------------------------------
// Problem constants (reference: B=2, S=2048, D=512, H=8, HD=64, D4=128)
// ---------------------------------------------------------------------------
#define BB   2
#define SS   2048
#define DD   512
#define HH   8
#define HD   64
#define D4   128
#define BS   (BB * SS)          // 4096 rows
#define BN_SCALE 0.9999950000374997f   // (1+1e-5)^-0.5

// Workspace layout (float offsets)
#define OFF_H1   ((size_t)0)           // [4096,2048]
#define OFF_H2   ((size_t)8388608)     // [4096,2048]
#define OFF_H3   ((size_t)0)           // [4096,128]  (reuse H1)
#define OFF_RVEC ((size_t)1048576)     // [4096]
#define OFF_WIN  ((size_t)1052672)     // 1 int
#define OFF_QP   ((size_t)2097152)     // [4096,512]
#define OFF_KP   ((size_t)4718592)     // [4096,64]
#define OFF_VP   ((size_t)5242880)     // [4096,64]
#define OFF_H4   ((size_t)8388608)     // [4096,512]  (reuse H2)
#define OFF_ATTN ((size_t)8388608)     // [4096,512]  (reuse H4 after window)
#define OFF_AP   ((size_t)10485760)    // [4096,512]
#define OFF_G    ((size_t)12582912)    // [4096,512]
#define OFF_CM   ((size_t)14680064)    // [1024]

// ---------------------------------------------------------------------------
// Feature gates (device pass only)
// ---------------------------------------------------------------------------
#if defined(__HIP_DEVICE_COMPILE__) && __has_builtin(__builtin_amdgcn_global_load_async_to_lds_b128)
#define USE_ASYNC_LDS 1
#else
#define USE_ASYNC_LDS 0
#endif

#define AS1 __attribute__((address_space(1)))
#define AS3 __attribute__((address_space(3)))

typedef int v4i_t __attribute__((__vector_size__(4 * sizeof(int))));

__device__ __forceinline__ AS1 v4i_t* as_global_v4(const void* p) {
  return (AS1 v4i_t*)(unsigned long long)p;
}
__device__ __forceinline__ AS3 v4i_t* as_lds_v4(const void* p) {
  return (AS3 v4i_t*)(unsigned)(unsigned long long)p;   // flat LDS addr low 32b
}
__device__ __forceinline__ void wait_async_le4() {
#if __has_builtin(__builtin_amdgcn_s_wait_asynccnt)
  __builtin_amdgcn_s_wait_asynccnt(4);
#else
  asm volatile("s_wait_asynccnt 0x4" ::: "memory");
#endif
}
__device__ __forceinline__ void wait_async_le0() {
#if __has_builtin(__builtin_amdgcn_s_wait_asynccnt)
  __builtin_amdgcn_s_wait_asynccnt(0);
#else
  asm volatile("s_wait_asynccnt 0x0" ::: "memory");
#endif
}

// ---------------------------------------------------------------------------
// WMMA types / helpers
// ---------------------------------------------------------------------------
typedef __attribute__((ext_vector_type(16))) __bf16 v16bf;
typedef __attribute__((ext_vector_type(8)))  float  v8f;

union FragBF { v16bf v; uint4 q[2]; };

__device__ __forceinline__ unsigned short f2bf(float f) {
  union { float f; unsigned u; } x; x.f = f;
  unsigned u = x.u;
  u += 0x7fffu + ((u >> 16) & 1u);     // round-to-nearest-even
  return (unsigned short)(u >> 16);
}
__device__ __forceinline__ unsigned pk_bf16(float lo, float hi) {
  return (unsigned)f2bf(lo) | ((unsigned)f2bf(hi) << 16);
}

__device__ __forceinline__ float silu_f(float v) {
  return v / (1.0f + __expf(-v));
}
__device__ __forceinline__ float gelu_f(float v) {
  return 0.5f * v * (1.0f + erff(v * 0.70710678118654752f));
}

#define ACT_NONE     0
#define ACT_SILU     1
#define ACT_GELU_BNS 2

// ---------------------------------------------------------------------------
// Generic WMMA GEMM: C[M,N] = act( [Cprev +] A[M,K] * W[N,K]^T + bias )
// Block 256 threads (8 waves), 64x64 tile, K-step 32, bf16 WMMA.
// Async double-buffered fp32 staging -> LDS when available; sync fallback.
// grid = (N/64, M/64)
// ---------------------------------------------------------------------------
__global__ __launch_bounds__(256) void gemm_wmma_kernel(
    const float* __restrict__ A, int lda,
    const float* __restrict__ W, int ldw, int koff,
    const float* __restrict__ bias,
    float* __restrict__ C, int ldc,
    int K, int act, int accum)
{
  __shared__ __align__(16) unsigned short As[64][32];
  __shared__ __align__(16) unsigned short Ws[64][32];

  const int tid  = threadIdx.x;
  const int lane = tid & 31;
  const int wave = tid >> 5;
  const int half = lane >> 4;
  const int l16  = lane & 15;
  const int mi   = wave & 3;           // m sub-tile (0..3)
  const int ni0  = (wave >> 2) * 2;    // two n sub-tiles per wave
  const int m0   = blockIdx.y * 64;
  const int n0   = blockIdx.x * 64;

  const int sm = tid >> 2;             // staging row 0..63
  const int sk = (tid & 3) * 8;        // staging k offset 0/8/16/24

  v8f acc0 = {};
  v8f acc1 = {};

#if USE_ASYNC_LDS
  __shared__ __align__(16) float Af[2][64][32];
  __shared__ __align__(16) float Wf[2][64][32];

  // prime the pipeline: tile k0=0 -> buffer 0 (4 async ops per thread)
  {
    const float* pa = A + (size_t)(m0 + sm) * lda + sk;
    const float* pw = W + (size_t)(n0 + sm) * ldw + koff + sk;
    __builtin_amdgcn_global_load_async_to_lds_b128(as_global_v4(pa),     as_lds_v4(&Af[0][sm][sk]),     0, 0);
    __builtin_amdgcn_global_load_async_to_lds_b128(as_global_v4(pa + 4), as_lds_v4(&Af[0][sm][sk + 4]), 0, 0);
    __builtin_amdgcn_global_load_async_to_lds_b128(as_global_v4(pw),     as_lds_v4(&Wf[0][sm][sk]),     0, 0);
    __builtin_amdgcn_global_load_async_to_lds_b128(as_global_v4(pw + 4), as_lds_v4(&Wf[0][sm][sk + 4]), 0, 0);
  }
  for (int k0 = 0; k0 < K; k0 += 32) {
    const int cur = (k0 >> 5) & 1;
    if (k0 + 32 < K) {                 // prefetch next tile, then retire current
      const int nxt = cur ^ 1;
      const float* pa = A + (size_t)(m0 + sm) * lda + (k0 + 32) + sk;
      const float* pw = W + (size_t)(n0 + sm) * ldw + koff + (k0 + 32) + sk;
      __builtin_amdgcn_global_load_async_to_lds_b128(as_global_v4(pa),     as_lds_v4(&Af[nxt][sm][sk]),     0, 0);
      __builtin_amdgcn_global_load_async_to_lds_b128(as_global_v4(pa + 4), as_lds_v4(&Af[nxt][sm][sk + 4]), 0, 0);
      __builtin_amdgcn_global_load_async_to_lds_b128(as_global_v4(pw),     as_lds_v4(&Wf[nxt][sm][sk]),     0, 0);
      __builtin_amdgcn_global_load_async_to_lds_b128(as_global_v4(pw + 4), as_lds_v4(&Wf[nxt][sm][sk + 4]), 0, 0);
      wait_async_le4();
    } else {
      wait_async_le0();
    }
    __syncthreads();
    // convert staged fp32 -> packed bf16 tiles (one ds_store_b128 per 8 elems)
    {
      const float4 a0 = *(const float4*)&Af[cur][sm][sk];
      const float4 a1 = *(const float4*)&Af[cur][sm][sk + 4];
      uint4 u;
      u.x = pk_bf16(a0.x, a0.y); u.y = pk_bf16(a0.z, a0.w);
      u.z = pk_bf16(a1.x, a1.y); u.w = pk_bf16(a1.z, a1.w);
      *(uint4*)&As[sm][sk] = u;
      const float4 b0 = *(const float4*)&Wf[cur][sm][sk];
      const float4 b1 = *(const float4*)&Wf[cur][sm][sk + 4];
      uint4 w4;
      w4.x = pk_bf16(b0.x, b0.y); w4.y = pk_bf16(b0.z, b0.w);
      w4.z = pk_bf16(b1.x, b1.y); w4.w = pk_bf16(b1.z, b1.w);
      *(uint4*)&Ws[sm][sk] = w4;
    }
    __syncthreads();
#else
  for (int k0 = 0; k0 < K; k0 += 32) {
    // ---- sync stage: load fp32, pack bf16, single b128 LDS store each ----
    {
      const float* pa = A + (size_t)(m0 + sm) * lda + k0 + sk;
      const float4 a0 = *(const float4*)(pa);
      const float4 a1 = *(const float4*)(pa + 4);
      uint4 u;
      u.x = pk_bf16(a0.x, a0.y); u.y = pk_bf16(a0.z, a0.w);
      u.z = pk_bf16(a1.x, a1.y); u.w = pk_bf16(a1.z, a1.w);
      *(uint4*)&As[sm][sk] = u;
      const float* pw = W + (size_t)(n0 + sm) * ldw + koff + k0 + sk;
      const float4 b0 = *(const float4*)(pw);
      const float4 b1 = *(const float4*)(pw + 4);
      uint4 w4;
      w4.x = pk_bf16(b0.x, b0.y); w4.y = pk_bf16(b0.z, b0.w);
      w4.z = pk_bf16(b1.x, b1.y); w4.w = pk_bf16(b1.z, b1.w);
      *(uint4*)&Ws[sm][sk] = w4;
    }
    __syncthreads();
#endif

    // ---- A fragment (16x32): lane m = l16; K pairs 8*half.., 16+8*half.. ----
    FragBF fa;
    {
      const unsigned short* r = &As[mi * 16 + l16][0];
      fa.q[0] = *(const uint4*)(r + 8 * half);
      fa.q[1] = *(const uint4*)(r + 16 + 8 * half);
    }
    // ---- B fragments (32x16): lane n = l16; K pairs 8*half.., 16+8*half.. ----
    FragBF fb0, fb1;
    {
      const unsigned short* r0 = &Ws[ni0 * 16 + l16][0];
      fb0.q[0] = *(const uint4*)(r0 + 8 * half);
      fb0.q[1] = *(const uint4*)(r0 + 16 + 8 * half);
      const unsigned short* r1 = &Ws[(ni0 + 1) * 16 + l16][0];
      fb1.q[0] = *(const uint4*)(r1 + 8 * half);
      fb1.q[1] = *(const uint4*)(r1 + 16 + 8 * half);
    }
    acc0 = __builtin_amdgcn_wmma_f32_16x16x32_bf16(false, fa.v, false, fb0.v,
                                                   (short)0, acc0, false, false);
    acc1 = __builtin_amdgcn_wmma_f32_16x16x32_bf16(false, fa.v, false, fb1.v,
                                                   (short)0, acc1, false, false);
    __syncthreads();
  }

  // ---- epilogue: C layout lane n = l16, VGPR j -> row j + 8*half ----
  for (int t = 0; t < 2; ++t) {
    v8f acc = t ? acc1 : acc0;
    const int col = n0 + (ni0 + t) * 16 + l16;
    const float bv = bias ? bias[col] : 0.0f;
    for (int j = 0; j < 8; ++j) {
      const int row = m0 + mi * 16 + j + 8 * half;
      float v = acc[j];
      if (accum) v += C[(size_t)row * ldc + col];
      v += bv;
      if (act == ACT_SILU)          v = silu_f(v);
      else if (act == ACT_GELU_BNS) v = gelu_f(v * BN_SCALE);
      C[(size_t)row * ldc + col] = v;
    }
  }
}

// ---------------------------------------------------------------------------
// Depthwise conv k=3, pad=1 over seq (per batch), then silu(x * BN_S).
// ---------------------------------------------------------------------------
__global__ __launch_bounds__(256) void dwconv3_kernel(
    const float* __restrict__ Hin, const float* __restrict__ wdw,
    float* __restrict__ Hout)
{
  const int idx = blockIdx.x * 256 + threadIdx.x;   // 4096*2048 total
  const int o  = idx & 2047;
  const int bs = idx >> 11;
  const int s  = bs & (SS - 1);
  const float w0 = wdw[o * 3 + 0], w1 = wdw[o * 3 + 1], w2 = wdw[o * 3 + 2];
  const float xm = (s > 0)      ? Hin[(size_t)(bs - 1) * 2048 + o] : 0.0f;
  const float xc =               Hin[(size_t)bs * 2048 + o];
  const float xp = (s < SS - 1) ? Hin[(size_t)(bs + 1) * 2048 + o] : 0.0f;
  const float v = (xm * w0 + xc * w1 + xp * w2) * BN_SCALE;
  Hout[idx] = silu_f(v);
}

// ---------------------------------------------------------------------------
// Per-channel scale (dw 1x1) then silu(x * BN_S), in place. [(b,s), 512]
// ---------------------------------------------------------------------------
__global__ __launch_bounds__(256) void scale_silu_kernel(
    float* __restrict__ X, const float* __restrict__ wdw2)
{
  const int idx = blockIdx.x * 256 + threadIdx.x;   // 4096*512
  const int o = idx & 511;
  const float v = X[idx] * wdw2[o] * BN_SCALE;
  X[idx] = silu_f(v);
}

// ---------------------------------------------------------------------------
// proj2 (N=1): r[row] = BN_S * dot(H[row, :512], wp2).  Wave per row.
// ---------------------------------------------------------------------------
__global__ __launch_bounds__(256) void rowdot_kernel(
    const float* __restrict__ Hin, const float* __restrict__ wp2,
    float* __restrict__ r)
{
  const int wave = threadIdx.x >> 5, lane = threadIdx.x & 31;
  const int row = blockIdx.x * 8 + wave;
  float s = 0.0f;
  for (int c = lane; c < 512; c += 32)
    s += Hin[(size_t)row * 512 + c] * wp2[c];
  for (int off = 1; off < 32; off <<= 1) s += __shfl_xor(s, off, 32);
  if (lane == 0) r[row] = s * BN_SCALE;
}

// ---------------------------------------------------------------------------
// Deterministic mean over 4096 values -> window int = min(int(3+ratio*29), S)
// ---------------------------------------------------------------------------
__global__ __launch_bounds__(1024) void winred_kernel(
    const float* __restrict__ r, int* __restrict__ win)
{
  __shared__ float sm[1024];
  const int t = threadIdx.x;
  sm[t] = r[t] + r[t + 1024] + r[t + 2048] + r[t + 3072];
  __syncthreads();
  for (int off = 512; off > 0; off >>= 1) {
    if (t < off) sm[t] += sm[t + off];
    __syncthreads();
  }
  if (t == 0) {
    const float ratio = sm[0] / 4096.0f;
    int w = (int)(3.0f + ratio * 29.0f);   // truncation like astype(int32)
    if (w > SS) w = SS;
    *win = w;
  }
}

// ---------------------------------------------------------------------------
// Flash-style MQA attention. One wave per (16 q-rows, head, batch).
// ---------------------------------------------------------------------------
__global__ __launch_bounds__(32) void attn_kernel(
    const float* __restrict__ Qp, const float* __restrict__ Kp,
    const float* __restrict__ Vp, const int* __restrict__ winp,
    float* __restrict__ Ob)
{
  __shared__ __align__(16) unsigned short Qs[16][64];
  __shared__ __align__(16) unsigned short Ks[32][64];
  __shared__ __align__(16) unsigned short Vts[64][32];   // transposed V chunk
  __shared__ __align__(16) unsigned short Ps[16][32];

  const int lane = threadIdx.x;
  const int l16  = lane & 15, half = lane >> 4;
  const int qt = blockIdx.x, h = blockIdx.y, b = blockIdx.z;
  const int qbase = qt * 16;

  // stage Q tile (16x64) as packed bf16
  {
    const int r  = lane >> 1;
    const int c0 = (lane & 1) * 32;
    const float* src = Qp + ((size_t)(b * SS + qbase + r)) * DD + h * HD + c0;
    unsigned* qrow = (unsigned*)&Qs[r][c0];
    for (int c = 0; c < 16; ++c) qrow[c] = pk_bf16(src[2 * c], src[2 * c + 1]);
  }
  __syncthreads();

  // Q A-fragments, contraction halves hd 0..31 and 32..63
  FragBF qa0, qa1;
  {
    const unsigned short* r = &Qs[l16][0];
    qa0.q[0] = *(const uint4*)(r + 8 * half);
    qa0.q[1] = *(const uint4*)(r + 16 + 8 * half);
    qa1.q[0] = *(const uint4*)(r + 32 + 8 * half);
    qa1.q[1] = *(const uint4*)(r + 48 + 8 * half);
  }

  const int  win = *winp;
  const int  w2  = win >> 1;
  const bool allow_all = (win >= SS);

  float mstat[8], lstat[8];
  v8f   Oacc[4];
  const v8f vzero = {};
  for (int j = 0; j < 8; ++j) { mstat[j] = -3.0e38f; lstat[j] = 0.0f; }
  for (int nt = 0; nt < 4; ++nt) Oacc[nt] = vzero;

  const int lo = qbase - w2, hi = qbase + 15 + w2;

  for (int kc = 0; kc < SS; kc += 32) {
    if (!allow_all) {                           // wave-uniform skip
      if (kc + 31 < lo || kc > hi) continue;
    }
    // prefetch next chunk's K/V rows (global_prefetch_b8)
    {
      const int kn = kc + 32;
      if (kn < SS && (allow_all || kn <= hi)) {
        __builtin_prefetch(Kp + ((size_t)(b * SS + kn + lane)) * HD, 0, 0);
        __builtin_prefetch(Vp + ((size_t)(b * SS + kn + lane)) * HD, 0, 0);
      }
    }
    // stage K chunk [32x64] (packed rows) and V^T chunk [64x32]
    {
      const float* ks = Kp + ((size_t)(b * SS + kc + lane)) * HD;
      const float* vs = Vp + ((size_t)(b * SS + kc + lane)) * HD;
      unsigned* krow = (unsigned*)&Ks[lane][0];
      for (int c = 0; c < 32; ++c) krow[c] = pk_bf16(ks[2 * c], ks[2 * c + 1]);
      for (int c = 0; c < 64; ++c) Vts[c][lane] = f2bf(vs[c]);
    }
    __syncthreads();

    for (int t = 0; t < 2; ++t) {               // two 16-key sub-tiles
      FragBF kb0, kb1;
      const unsigned short* r = &Ks[t * 16 + l16][0];
      kb0.q[0] = *(const uint4*)(r + 8 * half);
      kb0.q[1] = *(const uint4*)(r + 16 + 8 * half);
      kb1.q[0] = *(const uint4*)(r + 32 + 8 * half);
      kb1.q[1] = *(const uint4*)(r + 48 + 8 * half);

      v8f s = {};
      s = __builtin_amdgcn_wmma_f32_16x16x32_bf16(false, qa0.v, false, kb0.v,
                                                  (short)0, s, false, false);
      s = __builtin_amdgcn_wmma_f32_16x16x32_bf16(false, qa1.v, false, kb1.v,
                                                  (short)0, s, false, false);
      const int kg = kc + t * 16 + l16;
      for (int j = 0; j < 8; ++j) {
        float sv = s[j] * 0.125f;               // 1/sqrt(64)
        const int qg = qbase + j + 8 * half;
        const bool ok = allow_all || (kg >= qg - w2 && kg <= qg + w2);
        sv = ok ? sv : -1.0e9f;
        float mx = sv;                          // row max over 16 lanes
        for (int off = 1; off < 16; off <<= 1) mx = fmaxf(mx, __shfl_xor(mx, off, 32));
        const float nm = fmaxf(mstat[j], mx);
        const float alpha = __expf(mstat[j] - nm);
        const float p = __expf(sv - nm);
        float ps = p;
        for (int off = 1; off < 16; off <<= 1) ps += __shfl_xor(ps, off, 32);
        lstat[j] = lstat[j] * alpha + ps;
        mstat[j] = nm;
        for (int nt = 0; nt < 4; ++nt) Oacc[nt][j] *= alpha;
        Ps[j + 8 * half][t * 16 + l16] = f2bf(p);
      }
    }
    __syncthreads();

    // P (16x32) as A-fragment
    FragBF pa;
    {
      const unsigned short* r = &Ps[l16][0];
      pa.q[0] = *(const uint4*)(r + 8 * half);
      pa.q[1] = *(const uint4*)(r + 16 + 8 * half);
    }
    // P . V : four 16-wide HD tiles
    for (int nt = 0; nt < 4; ++nt) {
      FragBF vb;
      const unsigned short* r = &Vts[nt * 16 + l16][0];
      vb.q[0] = *(const uint4*)(r + 8 * half);
      vb.q[1] = *(const uint4*)(r + 16 + 8 * half);
      Oacc[nt] = __builtin_amdgcn_wmma_f32_16x16x32_bf16(false, pa.v, false, vb.v,
                                                         (short)0, Oacc[nt], false, false);
    }
    __syncthreads();
  }

  for (int j = 0; j < 8; ++j) {
    const float inv = (lstat[j] > 0.0f) ? 1.0f / lstat[j] : 0.0f;
    const int row = qbase + j + 8 * half;
    float* dst = Ob + ((size_t)(b * SS + row)) * DD + h * HD;
    for (int nt = 0; nt < 4; ++nt) dst[nt * 16 + l16] = Oacc[nt][j] * inv;
  }
}

// ---------------------------------------------------------------------------
// Gated residual + RMSNorm.  Wave per row of 512.
// ---------------------------------------------------------------------------
__global__ __launch_bounds__(256) void gated_rms_kernel(
    const float* __restrict__ q, const float* __restrict__ g,
    const float* __restrict__ a, const float* __restrict__ rmsw,
    float* __restrict__ out)
{
  const int wave = threadIdx.x >> 5, lane = threadIdx.x & 31;
  const int row = blockIdx.x * 8 + wave;
  const size_t base = (size_t)row * DD;
  float vals[16];
  float ss = 0.0f;
  for (int k = 0; k < 16; ++k) {
    const int c = lane + 32 * k;
    const float gv = g[base + c];
    const float v  = q[base + c] * gv + a[base + c] * (1.0f - gv);
    vals[k] = v;
    ss += v * v;
  }
  for (int off = 1; off < 32; off <<= 1) ss += __shfl_xor(ss, off, 32);
  const float r = rsqrtf(ss / 512.0f + 1e-6f);
  for (int k = 0; k < 16; ++k) {
    const int c = lane + 32 * k;
    out[base + c] = vals[k] * r * rmsw[c];
  }
}

// ---------------------------------------------------------------------------
// Column mean over seq dim per (b,d), then subtraction.
// ---------------------------------------------------------------------------
__global__ __launch_bounds__(256) void colmean_kernel(
    const float* __restrict__ x, float* __restrict__ cm)
{
  __shared__ float sm[256];
  const int bi = blockIdx.x;          // 0..1023
  const int b = bi >> 9, d = bi & 511;
  float s = 0.0f;
  for (int si = threadIdx.x; si < SS; si += 256)
    s += x[((size_t)(b * SS + si)) * DD + d];
  sm[threadIdx.x] = s;
  __syncthreads();
  for (int off = 128; off > 0; off >>= 1) {
    if (threadIdx.x < off) sm[threadIdx.x] += sm[threadIdx.x + off];
    __syncthreads();
  }
  if (threadIdx.x == 0) cm[bi] = sm[0] / (float)SS;
}

__global__ __launch_bounds__(256) void submean_kernel(
    float* __restrict__ x, const float* __restrict__ cm)
{
  const size_t idx = (size_t)blockIdx.x * 256 + threadIdx.x;  // 4096*512
  const int d = (int)(idx & 511);
  const int b = (int)(idx >> 20);     // 2048*512 = 1<<20 per batch
  x[idx] -= cm[b * 512 + d];
}

// ---------------------------------------------------------------------------
// Launcher
// ---------------------------------------------------------------------------
extern "C" void kernel_launch(void* const* d_in, const int* in_sizes, int n_in,
                              void* d_out, int out_size, void* d_ws, size_t ws_size,
                              hipStream_t stream) {
  (void)in_sizes; (void)n_in; (void)out_size; (void)ws_size;
  const float* query  = (const float*)d_in[0];
  const float* key    = (const float*)d_in[1];
  const float* value  = (const float*)d_in[2];
  const float* wp_e1  = (const float*)d_in[3];   // [2048,512]
  const float* wp_dw1 = (const float*)d_in[4];   // [2048,3]
  const float* wp_p1  = (const float*)d_in[5];   // [128,2048]
  const float* wp_e2  = (const float*)d_in[6];   // [512,128]
  const float* wp_dw2 = (const float*)d_in[7];   // [512,1]
  const float* wp_p2  = (const float*)d_in[8];   // [1,512]
  const float* wq = (const float*)d_in[9];   const float* bq = (const float*)d_in[10];
  const float* wk = (const float*)d_in[11];  const float* bk = (const float*)d_in[12];
  const float* wv = (const float*)d_in[13];  const float* bv = (const float*)d_in[14];
  const float* wo = (const float*)d_in[15];  const float* bo = (const float*)d_in[16];
  const float* wg = (const float*)d_in[17];  const float* bg = (const float*)d_in[18];
  const float* rms_w = (const float*)d_in[19];

  float* wsf  = (float*)d_ws;
  float* h1   = wsf + OFF_H1;
  float* h2   = wsf + OFF_H2;
  float* h3   = wsf + OFF_H3;
  float* rvec = wsf + OFF_RVEC;
  int*   winp = (int*)(wsf + OFF_WIN);
  float* Qp   = wsf + OFF_QP;
  float* Kp   = wsf + OFF_KP;
  float* Vp   = wsf + OFF_VP;
  float* h4   = wsf + OFF_H4;
  float* attn = wsf + OFF_ATTN;
  float* ap   = wsf + OFF_AP;
  float* gbuf = wsf + OFF_G;
  float* cm   = wsf + OFF_CM;
  float* outp = (float*)d_out;

  // 1) expand1: [4096,512] x [2048,512]^T -> h1, silu
  gemm_wmma_kernel<<<dim3(2048/64, BS/64), 256, 0, stream>>>(
      query, 512, wp_e1, 512, 0, nullptr, h1, 2048, 512, ACT_SILU, 0);
  // 2) depthwise k=3 + silu(BN)
  dwconv3_kernel<<<(BS * 2048) / 256, 256, 0, stream>>>(h1, wp_dw1, h2);
  // 3) proj1: [4096,2048] x [128,2048]^T -> h3, gelu(x*BN)
  gemm_wmma_kernel<<<dim3(128/64, BS/64), 256, 0, stream>>>(
      h2, 2048, wp_p1, 2048, 0, nullptr, h3, 128, 2048, ACT_GELU_BNS, 0);
  // 4) expand2: [4096,128] x [512,128]^T -> h4, silu
  gemm_wmma_kernel<<<dim3(512/64, BS/64), 256, 0, stream>>>(
      h3, 128, wp_e2, 128, 0, nullptr, h4, 512, 128, ACT_SILU, 0);
  // 5) per-channel scale + silu(BN), in place
  scale_silu_kernel<<<(BS * 512) / 256, 256, 0, stream>>>(h4, wp_dw2);
  // 6) proj2 (N=1) -> rvec
  rowdot_kernel<<<BS / 8, 256, 0, stream>>>(h4, wp_p2, rvec);
  // 7) window scalar
  winred_kernel<<<1, 1024, 0, stream>>>(rvec, winp);
  // 8-10) Q/K/V projections
  gemm_wmma_kernel<<<dim3(512/64, BS/64), 256, 0, stream>>>(
      query, 512, wq, 512, 0, bq, Qp, 512, 512, ACT_NONE, 0);
  gemm_wmma_kernel<<<dim3(64/64, BS/64), 256, 0, stream>>>(
      key, 512, wk, 512, 0, bk, Kp, 64, 512, ACT_NONE, 0);
  gemm_wmma_kernel<<<dim3(64/64, BS/64), 256, 0, stream>>>(
      value, 512, wv, 512, 0, bv, Vp, 64, 512, ACT_NONE, 0);
  // 11) flash attention with dynamic window
  attn_kernel<<<dim3(SS / 16, HH, BB), 32, 0, stream>>>(Qp, Kp, Vp, winp, attn);
  // 12) out projection
  gemm_wmma_kernel<<<dim3(512/64, BS/64), 256, 0, stream>>>(
      attn, 512, wo, 512, 0, bo, ap, 512, 512, ACT_NONE, 0);
  // 13-14) gate = silu(query*wgA^T + attn*wgB^T + bg)  (wg is [512,1024])
  gemm_wmma_kernel<<<dim3(512/64, BS/64), 256, 0, stream>>>(
      query, 512, wg, 1024, 0, nullptr, gbuf, 512, 512, ACT_NONE, 0);
  gemm_wmma_kernel<<<dim3(512/64, BS/64), 256, 0, stream>>>(
      ap, 512, wg, 1024, 512, bg, gbuf, 512, 512, ACT_SILU, 1);
  // 15) gated residual + RMSNorm -> d_out
  gated_rms_kernel<<<BS / 8, 256, 0, stream>>>(query, gbuf, ap, rms_w, outp);
  // 16-17) subtract seq-mean per (b, d)
  colmean_kernel<<<BB * DD, 256, 0, stream>>>(outp, cm);
  submean_kernel<<<(BS * 512) / 256, 256, 0, stream>>>(outp, cm);
}
